// MultiheadLatentAttention_13752485282227
// MI455X (gfx1250) — matrix operand order repaired and checked by reading, not compiled, not run on hardware
//
#include <hip/hip_runtime.h>
#include <hip/hip_bf16.h>

typedef unsigned short u16;
typedef __attribute__((ext_vector_type(16))) __bf16 bf16x16;
typedef __attribute__((ext_vector_type(8)))  float   f32x8;

union FragB { uint4 u[2]; bf16x16 v; };   // 32B: one WMMA 16-bit A/B operand
union FragF { f32x8 v; float f[8]; };     // 32B: one WMMA f32 C/D operand

__device__ __forceinline__ u16 f2bf(float f) {
  unsigned u = __float_as_uint(f);
  u += 0x7FFFu + ((u >> 16) & 1u);        // round-to-nearest-even
  return (u16)(u >> 16);
}
__device__ __forceinline__ float bf2f(u16 h) {
  return __uint_as_float(((unsigned)h) << 16);
}

// ---- CDNA5 async LDS copy (ASYNCcnt-tracked, bypasses VGPRs) ----
// GVS mode: 64-bit SGPR base + 32-bit per-lane byte offset; VDST VGPR holds LDS byte address.
__device__ __forceinline__ void async_ld_b128(unsigned lds_addr, const void* base, unsigned voff) {
  asm volatile("global_load_async_to_lds_b128 %0, %1, %2"
               :
               : "v"(lds_addr), "v"(voff), "s"(base)
               : "memory");
}
__device__ __forceinline__ void wait_async0() {
  asm volatile("s_wait_asynccnt 0x0" ::: "memory");
}

// ---------------------------------------------------------------- convert
__global__ __launch_bounds__(256) void cvt_f32_bf16_kernel(
    const float* __restrict__ src, u16* __restrict__ dst, size_t n4) {
  size_t i = (size_t)blockIdx.x * 256 + threadIdx.x;
  if (i >= n4) return;
  float4 f = ((const float4*)src)[i];
  uint2 o;
  o.x = (unsigned)f2bf(f.x) | ((unsigned)f2bf(f.y) << 16);
  o.y = (unsigned)f2bf(f.z) | ((unsigned)f2bf(f.w) << 16);
  ((uint2*)dst)[i] = o;
}

// ---------------------------------------------------------------- GEMM (bf16 x bf16 -> f32/bf16)
// C[M,N] = A[M,K] * B[K,N], row-major. Block tile 128x64, wave tile 32x32, BK=32.
// Double-buffered LDS; A tiles via global_load_async_to_lds_b128, B tiles via
// global->reg prefetch + transposed ds store while WMMAs of the current step run.
template <bool OUT_BF16>
__global__ __launch_bounds__(256) void gemm_bf16_kernel(
    const u16* __restrict__ A, const u16* __restrict__ B, void* __restrict__ Cv,
    int M, int N, int K) {
  __shared__ __align__(16) u16 sA[2][128 * 32];
  __shared__ __align__(16) u16 sBt[2][64 * 32];   // B^T tile: [n][k]

  const int tid  = threadIdx.x;
  const int lane = tid & 31, lm = lane & 15, lh = lane >> 4;
  const int wave = tid >> 5;
  const int wm = wave >> 1, wn = wave & 1;     // 4x2 wave grid
  const int bm = blockIdx.y * 128, bn = blockIdx.x * 64;

  const int brow = tid >> 3, bc8 = (tid & 7) * 8;  // B-staging slot (32 x 64)

  FragF acc[2][2];
  f32x8 z = {};
#pragma unroll
  for (int i = 0; i < 2; ++i)
#pragma unroll
    for (int j = 0; j < 2; ++j) acc[i][j].v = z;

  auto issueA = [&](int kt, int buf) {
#pragma unroll
    for (int it = 0; it < 2; ++it) {
      int slot = tid + it * 256;
      int row = slot >> 2, c8 = (slot & 3) * 8;
      unsigned lds = (unsigned)(size_t)(&sA[buf][row * 32 + c8]);
      unsigned off = (unsigned)(((bm + row) * K + kt + c8) * 2);
      async_ld_b128(lds, A, off);
    }
  };
  auto loadB = [&](int kt) {
    return *(const uint4*)(B + (size_t)(kt + brow) * N + bn + bc8);
  };
  auto storeB = [&](uint4 v, int buf) {
    union { uint4 q; u16 e[8]; } t;
    t.q = v;
#pragma unroll
    for (int j = 0; j < 8; ++j) sBt[buf][(bc8 + j) * 32 + brow] = t.e[j];
  };

  // prologue: stage k-step 0 into buffer 0
  issueA(0, 0);
  uint4 breg = loadB(0);
  storeB(breg, 0);
  wait_async0();
  __syncthreads();

  int cur = 0;
  for (int kt = 0; kt < K; kt += 32) {
    const int nxt = cur ^ 1;
    const bool more = (kt + 32) < K;
    if (more) {
      issueA(kt + 32, nxt);       // ASYNCcnt copy straight to LDS, overlapped with WMMA
      breg = loadB(kt + 32);      // B tile in flight during compute
      if (kt + 64 < K)
        __builtin_prefetch(B + (size_t)(kt + 64 + brow) * N + bn + bc8, 0, 0);
    }

    FragB af[2], bfr[2];
#pragma unroll
    for (int i = 0; i < 2; ++i) {
      const u16* p = &sA[cur][(wm * 32 + i * 16 + lm) * 32 + lh * 8]; // A: K {0-7,16-23}/{8-15,24-31}
      af[i].u[0] = *(const uint4*)p;
      af[i].u[1] = *(const uint4*)(p + 16);
    }
#pragma unroll
    for (int j = 0; j < 2; ++j) {
      const u16* p = &sBt[cur][(wn * 32 + j * 16 + lm) * 32 + lh * 8]; // B: K {0-7,16-23}/{8-15,24-31}
      bfr[j].u[0] = *(const uint4*)p;
      bfr[j].u[1] = *(const uint4*)(p + 16);
    }
#pragma unroll
    for (int i = 0; i < 2; ++i)
#pragma unroll
      for (int j = 0; j < 2; ++j)
        acc[i][j].v = __builtin_amdgcn_wmma_f32_16x16x32_bf16(
            false, af[i].v, false, bfr[j].v, (short)0, acc[i][j].v, false, false);

    if (more) storeB(breg, nxt);
    wait_async0();
    __syncthreads();
    cur = nxt;
  }

  float* Cf = (float*)Cv;
  u16*   Cb = (u16*)Cv;
#pragma unroll
  for (int i = 0; i < 2; ++i) {
    int row0 = bm + wm * 32 + i * 16 + lh * 8;
#pragma unroll
    for (int j = 0; j < 2; ++j) {
      int col = bn + wn * 32 + j * 16 + lm;
#pragma unroll
      for (int r = 0; r < 8; ++r) {
        size_t idx = (size_t)(row0 + r) * N + col;
        float val = acc[i][j].f[r];
        if (OUT_BF16) Cb[idx] = f2bf(val);
        else          Cf[idx] = val;
      }
    }
  }
}

// ---------------------------------------------------------------- RMSNorm (f32 in -> bf16 out)
__global__ __launch_bounds__(256) void rmsnorm_bf16_kernel(
    const float* __restrict__ X, const float* __restrict__ W,
    u16* __restrict__ Y, int L) {
  __shared__ float red[256];
  int row = blockIdx.x, tid = threadIdx.x;
  const float* x = X + (size_t)row * L;
  float s = 0.f;
  for (int i = tid; i < L; i += 256) { float v = x[i]; s += v * v; }
  red[tid] = s; __syncthreads();
  for (int off = 128; off > 0; off >>= 1) {
    if (tid < off) red[tid] += red[tid + off];
    __syncthreads();
  }
  float scale = rsqrtf(red[0] / (float)L + 1e-6f);
  u16* y = Y + (size_t)row * L;
  for (int i = tid; i < L; i += 256) y[i] = f2bf(x[i] * scale * W[i]);
}

// ---------------------------------------------------------------- RoPE
// q: [4096][16][192] bf16, rotate dims [128,192). half-pairs (i, 32+i), dim=64.
__global__ __launch_bounds__(512) void rope_q_kernel(u16* __restrict__ Q) {
  constexpr int T = 2048;
  int row = blockIdx.x;
  int h = threadIdx.x >> 5;
  int i = threadIdx.x & 31;
  int t = row & (T - 1);
  u16* base = Q + ((size_t)row * 16 + h) * 192 + 128;
  float x1 = bf2f(base[i]), x2 = bf2f(base[32 + i]);
  float inv = expf(-((float)i * (1.0f / 32.0f)) * 9.210340371976184f); // theta^(-2i/64)
  float f = (float)t * inv;
  float c = cosf(f), s = sinf(f);
  base[i]      = f2bf(x1 * c - x2 * s);
  base[32 + i] = f2bf(x2 * c + x1 * s);
}

// k_rope: [4096][64] f32 -> rope -> bf16
__global__ __launch_bounds__(32) void rope_k_kernel(
    const float* __restrict__ Kin, u16* __restrict__ Kout) {
  constexpr int T = 2048;
  int row = blockIdx.x, i = threadIdx.x;
  int t = row & (T - 1);
  const float* in = Kin + (size_t)row * 64;
  u16* outp = Kout + (size_t)row * 64;
  float x1 = in[i], x2 = in[32 + i];
  float inv = expf(-((float)i * (1.0f / 32.0f)) * 9.210340371976184f);
  float f = (float)t * inv;
  float c = cosf(f), s = sinf(f);
  outp[i]      = f2bf(x1 * c - x2 * s);
  outp[32 + i] = f2bf(x2 * c + x1 * s);
}

// ---------------------------------------------------------------- causal flash attention
// grid: (T/64, B*H); block: 128 threads (4 waves, wave w owns q rows [qb*64+16w, +16)).
// Q: [4096][16][192] bf16; KV: [4096][16][256] bf16 (128 k_pass + 128 v);
// KR: [4096][64] bf16 (shared across heads); O: [4096][16*128] bf16.
__global__ __launch_bounds__(128) void mla_attn_kernel(
    const u16* __restrict__ Q, const u16* __restrict__ KV,
    const u16* __restrict__ KR, u16* __restrict__ O) {
  constexpr int T = 2048;
  __shared__ __align__(16) u16 sK[64 * 192];    // K tile, row-major [kr][d]
  __shared__ __align__(16) u16 sVt[128 * 64];   // V^T tile [d][kr]
  __shared__ __align__(16) u16 sP[4 * 16 * 64]; // per-wave P staging

  const int tid = threadIdx.x, lane = tid & 31, lm = lane & 15, lh = lane >> 4;
  const int wave = tid >> 5;
  const int qb = blockIdx.x, bh = blockIdx.y;
  const int b = bh >> 4, h = bh & 15;
  const int qr0 = qb * 64 + wave * 16;

  // Q fragments for all 6 k-steps, kept in registers
  FragB qa[6];
  {
    const u16* qrow = Q + (((size_t)(b * T) + qr0 + lm) * 16 + h) * 192;
#pragma unroll
    for (int ks = 0; ks < 6; ++ks) {
      int k0 = ks * 32 + lh * 8;
      qa[ks].u[0] = *(const uint4*)(qrow + k0);
      qa[ks].u[1] = *(const uint4*)(qrow + k0 + 16);
    }
  }

  FragF oacc[8];
  f32x8 z = {};
#pragma unroll
  for (int n = 0; n < 8; ++n) oacc[n].v = z;
  float msta[8], lsta[8];
#pragma unroll
  for (int r = 0; r < 8; ++r) { msta[r] = -3.0e38f; lsta[r] = 0.f; }

  const float scale = 0.07216878364870323f;  // 1/sqrt(192)

  for (int kb = 0; kb <= qb; ++kb) {
    // stage K tile 64x192 via async-to-LDS, chunk-major so the KV/KR source
    // select is wave-uniform (uniform "s" base per executed branch).
    // slot = c*64 + krow; c in [0,24): c<16 -> k_pass from KV, c>=16 -> k_rope from KR.
#pragma unroll
    for (int it = 0; it < 12; ++it) {
      int slot = tid + it * 128;
      int c = slot >> 6, krow = slot & 63;
      int d0 = c * 8;
      int kt = kb * 64 + krow;
      unsigned lds = (unsigned)(size_t)(&sK[krow * 192 + d0]);
      if (d0 < 128) {
        unsigned off = (unsigned)((((unsigned)(b * T + kt) * 16 + h) * 256 + d0) * 2);
        async_ld_b128(lds, KV, off);
      } else {
        unsigned off = (unsigned)(((unsigned)(b * T + kt) * 64 + (d0 - 128)) * 2);
        async_ld_b128(lds, KR, off);
      }
    }
    // stage V^T 128x64 (element permute must pass through VGPRs)
#pragma unroll
    for (int it = 0; it < 8; ++it) {
      int slot = tid + it * 128;          // 64 rows x 16 chunks
      int krow = slot >> 4;
      int d0 = (slot & 15) * 8;
      int kt = kb * 64 + krow;
      uint4 v = *(const uint4*)(KV + (((size_t)(b * T) + kt) * 16 + h) * 256 + 128 + d0);
      union { uint4 q; u16 e[8]; } t; t.q = v;
#pragma unroll
      for (int j = 0; j < 8; ++j) sVt[(d0 + j) * 64 + krow] = t.e[j];
    }
    // prefetch next iteration's K/V and KR tiles (cacheline granularity)
    if (kb < qb) {
#pragma unroll
      for (int pi = 0; pi < 2; ++pi) {
        int s2 = tid + pi * 128;                 // 64 rows x 4 x 64-elem chunks
        int krow = s2 >> 2, q4 = (s2 & 3) * 64;
        int kt = (kb + 1) * 64 + krow;
        __builtin_prefetch(KV + (((size_t)(b * T) + kt) * 16 + h) * 256 + q4, 0, 0);
      }
      if (tid < 64)
        __builtin_prefetch(KR + ((size_t)(b * T) + (kb + 1) * 64 + tid) * 64, 0, 0);
    }
    wait_async0();
    __syncthreads();

    // S = Q K^T  (16x64 per wave, 4 n-tiles, 6 k-steps)
    FragF s[4];
#pragma unroll
    for (int j = 0; j < 4; ++j) s[j].v = z;
#pragma unroll
    for (int ks = 0; ks < 6; ++ks) {
#pragma unroll
      for (int j = 0; j < 4; ++j) {
        FragB kf;
        const u16* p = sK + (j * 16 + lm) * 192 + ks * 32 + lh * 8;
        kf.u[0] = *(const uint4*)p;
        kf.u[1] = *(const uint4*)(p + 16);
        s[j].v = __builtin_amdgcn_wmma_f32_16x16x32_bf16(
            false, qa[ks].v, false, kf.v, (short)0, s[j].v, false, false);
      }
    }

    // scale + causal mask (diagonal block only)
    const bool diag = (kb == qb);
#pragma unroll
    for (int j = 0; j < 4; ++j) {
      int col = kb * 64 + j * 16 + lm;
#pragma unroll
      for (int r = 0; r < 8; ++r) {
        float v = s[j].f[r] * scale;
        if (diag && col > qr0 + lh * 8 + r) v = -3.0e38f;
        s[j].f[r] = v;
      }
    }

    // online softmax: rows live in 16-lane half-wave groups
#pragma unroll
    for (int r = 0; r < 8; ++r) {
      float m = fmaxf(fmaxf(s[0].f[r], s[1].f[r]), fmaxf(s[2].f[r], s[3].f[r]));
#pragma unroll
      for (int x = 8; x >= 1; x >>= 1) m = fmaxf(m, __shfl_xor(m, x, 32));
      float mn = fmaxf(msta[r], m);
      float alpha = __expf(msta[r] - mn);
      float rs = 0.f;
#pragma unroll
      for (int j = 0; j < 4; ++j) {
        float p = __expf(s[j].f[r] - mn);
        s[j].f[r] = p;
        rs += p;
      }
#pragma unroll
      for (int x = 8; x >= 1; x >>= 1) rs += __shfl_xor(rs, x, 32);
      lsta[r] = lsta[r] * alpha + rs;
      msta[r] = mn;
#pragma unroll
      for (int n = 0; n < 8; ++n) oacc[n].f[r] *= alpha;
    }

    // P: C-layout -> A-layout via per-wave LDS staging
    u16* pw = sP + wave * 16 * 64;
#pragma unroll
    for (int j = 0; j < 4; ++j)
#pragma unroll
      for (int r = 0; r < 8; ++r)
        pw[(lh * 8 + r) * 64 + j * 16 + lm] = f2bf(s[j].f[r]);

    // O += P * V  (2 k-steps over 64 kv rows, 8 n-tiles over d_v=128)
#pragma unroll
    for (int ks2 = 0; ks2 < 2; ++ks2) {
      FragB pa;
      const u16* pp = pw + lm * 64 + ks2 * 32 + lh * 8;
      pa.u[0] = *(const uint4*)pp;
      pa.u[1] = *(const uint4*)(pp + 16);
#pragma unroll
      for (int n = 0; n < 8; ++n) {
        FragB vb;
        const u16* vp = sVt + (n * 16 + lm) * 64 + ks2 * 32 + lh * 8;
        vb.u[0] = *(const uint4*)vp;
        vb.u[1] = *(const uint4*)(vp + 16);
        oacc[n].v = __builtin_amdgcn_wmma_f32_16x16x32_bf16(
            false, pa.v, false, vb.v, (short)0, oacc[n].v, false, false);
      }
    }
    __syncthreads();
  }

  // epilogue: O /= l, store bf16 [row][h*128 + d]
#pragma unroll
  for (int r = 0; r < 8; ++r) {
    int row = qr0 + lh * 8 + r;
    float inv = 1.f / lsta[r];
    u16* orow = O + ((size_t)(b * T) + row) * 2048 + h * 128;
#pragma unroll
    for (int n = 0; n < 8; ++n)
      orow[n * 16 + lm] = f2bf(oacc[n].f[r] * inv);
  }
}

// ---------------------------------------------------------------- launch
extern "C" void kernel_launch(void* const* d_in, const int* in_sizes, int n_in,
                              void* d_out, int out_size, void* d_ws, size_t ws_size,
                              hipStream_t stream) {
  (void)in_sizes; (void)n_in; (void)out_size; (void)ws_size;
  const float* hs    = (const float*)d_in[0];
  const float* Wq_a  = (const float*)d_in[1];
  const float* q_nw  = (const float*)d_in[2];
  const float* Wq_b  = (const float*)d_in[3];
  const float* Wk_r  = (const float*)d_in[4];
  const float* Wkv_a = (const float*)d_in[5];
  const float* kv_nw = (const float*)d_in[6];
  const float* Wkv_b = (const float*)d_in[7];
  const float* Wo    = (const float*)d_in[8];
  float* out = (float*)d_out;

  constexpr int B = 2, T = 2048, D = 2048, Hn = 16;
  constexpr int QL = 1536, KVL = 512;
  const int M = B * T;  // 4096

  char* p = (char*)d_ws;
  auto alloc = [&](size_t bytes) {
    char* r = p;
    p += (bytes + 255) & ~(size_t)255;
    return r;
  };
  u16*   hs_b   = (u16*)alloc((size_t)M * D * 2);
  u16*   wqa_b  = (u16*)alloc((size_t)D * QL * 2);
  u16*   wqb_b  = (u16*)alloc((size_t)QL * Hn * 192 * 2);
  u16*   wkr_b  = (u16*)alloc((size_t)D * 64 * 2);
  u16*   wkva_b = (u16*)alloc((size_t)D * KVL * 2);
  u16*   wkvb_b = (u16*)alloc((size_t)KVL * Hn * 256 * 2);
  u16*   wo_b   = (u16*)alloc((size_t)Hn * 128 * D * 2);
  float* qa_f   = (float*)alloc((size_t)M * QL * 4);
  float* kva_f  = (float*)alloc((size_t)M * KVL * 4);
  float* kr_f   = (float*)alloc((size_t)M * 64 * 4);
  u16*   qn_b   = (u16*)alloc((size_t)M * QL * 2);
  u16*   kvn_b  = (u16*)alloc((size_t)M * KVL * 2);
  u16*   q_b    = (u16*)alloc((size_t)M * Hn * 192 * 2);
  u16*   kv_b   = (u16*)alloc((size_t)M * Hn * 256 * 2);
  u16*   kr_b   = (u16*)alloc((size_t)M * 64 * 2);
  u16*   o_b    = (u16*)alloc((size_t)M * Hn * 128 * 2);

  auto cvt = [&](const float* s, u16* d, size_t n) {
    size_t n4 = n / 4;
    cvt_f32_bf16_kernel<<<dim3((unsigned)((n4 + 255) / 256)), dim3(256), 0, stream>>>(s, d, n4);
  };
  cvt(hs, hs_b, (size_t)M * D);
  cvt(Wq_a, wqa_b, (size_t)D * QL);
  cvt(Wq_b, wqb_b, (size_t)QL * Hn * 192);
  cvt(Wk_r, wkr_b, (size_t)D * 64);
  cvt(Wkv_a, wkva_b, (size_t)D * KVL);
  cvt(Wkv_b, wkvb_b, (size_t)KVL * Hn * 256);
  cvt(Wo, wo_b, (size_t)Hn * 128 * D);

  auto gemm_f32 = [&](const u16* A, const u16* Bm, float* C, int m, int n, int k) {
    gemm_bf16_kernel<false><<<dim3(n / 64, m / 128), dim3(256), 0, stream>>>(A, Bm, (void*)C, m, n, k);
  };
  auto gemm_b16 = [&](const u16* A, const u16* Bm, u16* C, int m, int n, int k) {
    gemm_bf16_kernel<true><<<dim3(n / 64, m / 128), dim3(256), 0, stream>>>(A, Bm, (void*)C, m, n, k);
  };

  gemm_f32(hs_b, wqa_b,  qa_f,  M, QL,  D);
  gemm_f32(hs_b, wkva_b, kva_f, M, KVL, D);
  gemm_f32(hs_b, wkr_b,  kr_f,  M, 64,  D);

  rmsnorm_bf16_kernel<<<dim3(M), dim3(256), 0, stream>>>(qa_f,  q_nw,  qn_b,  QL);
  rmsnorm_bf16_kernel<<<dim3(M), dim3(256), 0, stream>>>(kva_f, kv_nw, kvn_b, KVL);

  gemm_b16(qn_b,  wqb_b,  q_b,  M, Hn * 192, QL);
  gemm_b16(kvn_b, wkvb_b, kv_b, M, Hn * 256, KVL);

  rope_q_kernel<<<dim3(M), dim3(512), 0, stream>>>(q_b);
  rope_k_kernel<<<dim3(M), dim3(32), 0, stream>>>(kr_f, kr_b);

  mla_attn_kernel<<<dim3(T / 64, B * Hn), dim3(128), 0, stream>>>(q_b, kv_b, kr_b, o_b);

  gemm_f32(o_b, wo_b, out, M, D, Hn * 128);
}
